// DeviceAwareMoELayer_21792664059953
// MI455X (gfx1250) — matrix operand order repaired and checked by reading, not compiled, not run on hardware
//
#include <hip/hip_runtime.h>

typedef __attribute__((ext_vector_type(16))) __bf16 v16bf;
typedef __attribute__((ext_vector_type(8)))  __bf16 v8bf;
typedef __attribute__((ext_vector_type(8)))  float  v8f;

#define T_TOK 2048
#define DIM   1024
#define HID   2048
#define NEXP  16
#define ROWS_PER_BLOCK 64
#define MT (T_TOK / ROWS_PER_BLOCK)   // 32 row-tiles max per expert

// ---------------- workspace layout ----------------
// [0, 256)                  : int counts[64]          (only NEXP used)
// [256, +NEXP*T*4)          : int lists[NEXP][T]
// [.., +(T+1)*HID*2)        : __bf16 hbuf[T+1][HID]   (row T = dummy sink)
// [.., +DIM*4)              : float outdummy[DIM]     (sink for padded rows)

__global__ void moe_zero_counts(int* __restrict__ counts) {
    if (threadIdx.x < NEXP) counts[threadIdx.x] = 0;
}

// One block per token: 256 threads = 16 experts x 16 D-chunks.
__global__ void __launch_bounds__(256)
moe_gate(const float* __restrict__ x, const float* __restrict__ gw,
         const float* __restrict__ gb, int* __restrict__ counts,
         int* __restrict__ lists) {
    const int t   = blockIdx.x;
    const int tid = threadIdx.x;
    const int e     = tid & 15;
    const int chunk = tid >> 4;
    const float* __restrict__ xr = x + (size_t)t * DIM;

    float acc = 0.0f;
    const int d0 = chunk * (DIM / 16);
    #pragma unroll 4
    for (int d = d0; d < d0 + DIM / 16; ++d)
        acc += xr[d] * gw[d * NEXP + e];

    __shared__ float part[16][17];
    __shared__ float logits[16];
    part[chunk][e] = acc;
    __syncthreads();

    if (tid < NEXP) {
        float s = gb[tid];
        #pragma unroll
        for (int c = 0; c < 16; ++c) s += part[c][tid];
        logits[tid] = s;
    }
    __syncthreads();

    if (tid == 0) {
        int   best = 0;
        float bv   = logits[0];
        #pragma unroll
        for (int i = 1; i < NEXP; ++i) {
            if (logits[i] > bv) { bv = logits[i]; best = i; }   // first-max wins
        }
        int pos = atomicAdd(&counts[best], 1);
        lists[best * T_TOK + pos] = t;
    }
}

// A-fragment (16x32 bf16 layout) from an fp32 row: two aligned float4 pairs.
__device__ __forceinline__ v16bf load_a_f32(const float* __restrict__ row, int hi) {
    const float4* pa = (const float4*)(row + hi * 8);
    const float4* pb = (const float4*)(row + 16 + hi * 8);
    float4 f0 = pa[0], f1 = pa[1], f2 = pb[0], f3 = pb[1];
    v16bf a;
    a[0]  = (__bf16)f0.x; a[1]  = (__bf16)f0.y; a[2]  = (__bf16)f0.z; a[3]  = (__bf16)f0.w;
    a[4]  = (__bf16)f1.x; a[5]  = (__bf16)f1.y; a[6]  = (__bf16)f1.z; a[7]  = (__bf16)f1.w;
    a[8]  = (__bf16)f2.x; a[9]  = (__bf16)f2.y; a[10] = (__bf16)f2.z; a[11] = (__bf16)f2.w;
    a[12] = (__bf16)f3.x; a[13] = (__bf16)f3.y; a[14] = (__bf16)f3.z; a[15] = (__bf16)f3.w;
    return a;
}

// A-fragment from a bf16 row: two aligned 16B vector loads.
__device__ __forceinline__ v16bf load_a_bf16(const __bf16* __restrict__ row, int hi) {
    v8bf lo  = *(const v8bf*)(row + hi * 8);
    v8bf hi8 = *(const v8bf*)(row + 16 + hi * 8);
    v16bf a;
    #pragma unroll
    for (int i = 0; i < 8; ++i) { a[i] = lo[i]; a[8 + i] = hi8[i]; }
    return a;
}

// FFN layer 1: h[tok, :] = relu(x[tok, :] @ w1[e] + b1[e]), h stored bf16 by token id.
__global__ void __launch_bounds__(256)
moe_ffn1(const float* __restrict__ x, const float* __restrict__ w1,
         const float* __restrict__ b1, const int* __restrict__ counts,
         const int* __restrict__ lists, __bf16* __restrict__ hbuf) {
    const int e  = blockIdx.x / MT;
    const int mt = blockIdx.x % MT;
    const int count = counts[e];
    if (mt * ROWS_PER_BLOCK >= count) return;

    __shared__ int toks[ROWS_PER_BLOCK];
    const int tid = threadIdx.x;
    if (tid < ROWS_PER_BLOCK) {
        const int idx = mt * ROWS_PER_BLOCK + tid;
        toks[tid] = (idx < count) ? lists[e * T_TOK + idx] : -1;
    }
    __syncthreads();

    const int wv   = tid >> 5;
    const int lane = tid & 31;
    const int m16  = lane & 15;
    const int hi   = lane >> 4;
    const float* __restrict__ W = w1 + (size_t)e * DIM * HID;

    // Branchless A-row pointers: padded rows read token 0 (results discarded).
    const float* arow[4];
    #pragma unroll
    for (int rt = 0; rt < 4; ++rt) {
        const int tk = toks[rt * 16 + m16];
        arow[rt] = x + (size_t)(tk >= 0 ? tk : 0) * DIM;
    }
    // Store rows: padded rows -> dummy row T_TOK of hbuf.
    int srow[4][8];
    #pragma unroll
    for (int rt = 0; rt < 4; ++rt)
        #pragma unroll
        for (int v = 0; v < 8; ++v) {
            const int tk = toks[rt * 16 + hi * 8 + v];
            srow[rt][v] = (tk >= 0) ? tk : T_TOK;
        }

    for (int nt = wv; nt < HID / 16; nt += 8) {
        const int h0 = nt * 16;
        v8f c0 = {}, c1 = {}, c2 = {}, c3 = {};
        for (int k0 = 0; k0 < DIM; k0 += 32) {
            v16bf bfr;
            #pragma unroll
            for (int i = 0; i < 16; ++i) {
                const int k = (i < 8) ? (hi * 8 + i) : (16 + hi * 8 + (i - 8));                       // B 32x16 layout
                bfr[i] = (__bf16)W[(size_t)(k0 + k) * HID + h0 + m16];
            }
            v16bf a0 = load_a_f32(arow[0] + k0, hi);
            v16bf a1 = load_a_f32(arow[1] + k0, hi);
            v16bf a2 = load_a_f32(arow[2] + k0, hi);
            v16bf a3 = load_a_f32(arow[3] + k0, hi);
            c0 = __builtin_amdgcn_wmma_f32_16x16x32_bf16(false, a0, false, bfr, (short)0, c0, false, false);
            c1 = __builtin_amdgcn_wmma_f32_16x16x32_bf16(false, a1, false, bfr, (short)0, c1, false, false);
            c2 = __builtin_amdgcn_wmma_f32_16x16x32_bf16(false, a2, false, bfr, (short)0, c2, false, false);
            c3 = __builtin_amdgcn_wmma_f32_16x16x32_bf16(false, a3, false, bfr, (short)0, c3, false, false);
        }
        const float bias = b1[e * HID + h0 + m16];
        v8f cs[4] = {c0, c1, c2, c3};
        #pragma unroll
        for (int rt = 0; rt < 4; ++rt)
            #pragma unroll
            for (int v = 0; v < 8; ++v) {
                float val = fmaxf(cs[rt][v] + bias, 0.0f);
                hbuf[(size_t)srow[rt][v] * HID + h0 + m16] = (__bf16)val;
            }
    }
}

// FFN layer 2: out[tok, :] = h[tok, :] @ w2[e] + b2[e], fp32 scatter to d_out.
__global__ void __launch_bounds__(256)
moe_ffn2(const __bf16* __restrict__ hbuf, const float* __restrict__ w2,
         const float* __restrict__ b2, const int* __restrict__ counts,
         const int* __restrict__ lists, float* __restrict__ out,
         float* __restrict__ odummy) {
    const int e  = blockIdx.x / MT;
    const int mt = blockIdx.x % MT;
    const int count = counts[e];
    if (mt * ROWS_PER_BLOCK >= count) return;

    __shared__ int toks[ROWS_PER_BLOCK];
    const int tid = threadIdx.x;
    if (tid < ROWS_PER_BLOCK) {
        const int idx = mt * ROWS_PER_BLOCK + tid;
        toks[tid] = (idx < count) ? lists[e * T_TOK + idx] : -1;
    }
    __syncthreads();

    const int wv   = tid >> 5;
    const int lane = tid & 31;
    const int m16  = lane & 15;
    const int hi   = lane >> 4;
    const float* __restrict__ W = w2 + (size_t)e * HID * DIM;

    const __bf16* arow[4];
    #pragma unroll
    for (int rt = 0; rt < 4; ++rt) {
        const int tk = toks[rt * 16 + m16];
        arow[rt] = hbuf + (size_t)(tk >= 0 ? tk : 0) * HID;
    }
    int stok[4][8];
    #pragma unroll
    for (int rt = 0; rt < 4; ++rt)
        #pragma unroll
        for (int v = 0; v < 8; ++v)
            stok[rt][v] = toks[rt * 16 + hi * 8 + v];

    for (int nt = wv; nt < DIM / 16; nt += 8) {
        const int d0 = nt * 16;
        v8f c0 = {}, c1 = {}, c2 = {}, c3 = {};
        for (int k0 = 0; k0 < HID; k0 += 32) {
            v16bf bfr;
            #pragma unroll
            for (int i = 0; i < 16; ++i) {
                const int k = (i < 8) ? (hi * 8 + i) : (16 + hi * 8 + (i - 8));
                bfr[i] = (__bf16)W[(size_t)(k0 + k) * DIM + d0 + m16];
            }
            v16bf a0 = load_a_bf16(arow[0] + k0, hi);
            v16bf a1 = load_a_bf16(arow[1] + k0, hi);
            v16bf a2 = load_a_bf16(arow[2] + k0, hi);
            v16bf a3 = load_a_bf16(arow[3] + k0, hi);
            c0 = __builtin_amdgcn_wmma_f32_16x16x32_bf16(false, a0, false, bfr, (short)0, c0, false, false);
            c1 = __builtin_amdgcn_wmma_f32_16x16x32_bf16(false, a1, false, bfr, (short)0, c1, false, false);
            c2 = __builtin_amdgcn_wmma_f32_16x16x32_bf16(false, a2, false, bfr, (short)0, c2, false, false);
            c3 = __builtin_amdgcn_wmma_f32_16x16x32_bf16(false, a3, false, bfr, (short)0, c3, false, false);
        }
        const float bias = b2[e * DIM + d0 + m16];
        v8f cs[4] = {c0, c1, c2, c3};
        #pragma unroll
        for (int rt = 0; rt < 4; ++rt)
            #pragma unroll
            for (int v = 0; v < 8; ++v) {
                const int tk = stok[rt][v];
                float* __restrict__ p = (tk >= 0) ? (out + (size_t)tk * DIM) : odummy;
                p[d0 + m16] = cs[rt][v] + bias;
            }
    }
}

extern "C" void kernel_launch(void* const* d_in, const int* in_sizes, int n_in,
                              void* d_out, int out_size, void* d_ws, size_t ws_size,
                              hipStream_t stream) {
    const float* x  = (const float*)d_in[0];
    const float* gw = (const float*)d_in[1];
    const float* gb = (const float*)d_in[2];
    const float* w1 = (const float*)d_in[3];
    const float* b1 = (const float*)d_in[4];
    const float* w2 = (const float*)d_in[5];
    const float* b2 = (const float*)d_in[6];
    float* out = (float*)d_out;

    int*    counts  = (int*)d_ws;
    int*    lists   = counts + 64;                                  // 256B in
    __bf16* hbuf    = (__bf16*)(lists + NEXP * T_TOK);
    float*  odummy  = (float*)(hbuf + (size_t)(T_TOK + 1) * HID);

    moe_zero_counts<<<1, 32, 0, stream>>>(counts);
    moe_gate<<<T_TOK, 256, 0, stream>>>(x, gw, gb, counts, lists);
    moe_ffn1<<<NEXP * MT, 256, 0, stream>>>(x, w1, b1, counts, lists, hbuf);
    moe_ffn2<<<NEXP * MT, 256, 0, stream>>>(hbuf, w2, b2, counts, lists, out, odummy);
}
